// SoftAlignLocalAttention_23373212025420
// MI455X (gfx1250) — matrix operand order repaired and checked by reading, not compiled, not run on hardware
//
#include <hip/hip_runtime.h>
#include <hip/hip_bf16.h>
#include <stdint.h>

#define DEV_INLINE __device__ __forceinline__

typedef __attribute__((ext_vector_type(16))) _Float16 v16h;
typedef __attribute__((ext_vector_type(8)))  float    v8f;
typedef __attribute__((ext_vector_type(4)))  unsigned int u32x4;
typedef __attribute__((ext_vector_type(8)))  int      i32x8;
typedef __attribute__((ext_vector_type(4)))  int      i32x4;

#if defined(__has_builtin)
#if __has_builtin(__builtin_amdgcn_tensor_load_to_lds) && \
    __has_builtin(__builtin_amdgcn_s_wait_tensorcnt)
#define USE_TDM 1
#endif
#endif
#ifndef USE_TDM
#define USE_TDM 0
#endif

namespace {

constexpr int B_  = 4;
constexpr int C_  = 512;
constexpr int T_  = 4096;
constexpr int H_  = 8;
constexpr int HD_ = 64;
constexpr float SCALE = 0.125f;     // 1/sqrt(64)

constexpr int ASTR = 520;           // A panel LDS row stride (halves): 512 + 8 pad
constexpr int BSTR = 40;            // B tile  LDS row stride (halves): 32 + 8 pad

union HFrag  { v16h v; uint32_t u[8]; uint4 q[2]; _Float16 h[16]; };
union OutPk  { _Float16 h[8]; uint4 q; };

DEV_INLINE v8f zero8() { v8f z; for (int i = 0; i < 8; ++i) z[i] = 0.0f; return z; }

DEV_INLINE v8f wmma16(v16h a, v16h b, v8f c) {
  return __builtin_amdgcn_wmma_f32_16x16x32_f16(false, a, false, b, (short)0, c, false, false);
}

#if USE_TDM
// TDM: DMA a 64-row x 512-half weight panel into LDS with 4-DWORD padding
// every 256 DWORDs (=> 520-half LDS row stride). D# per CDNA5 ISA.
// 6-arg builtin form (clang-23 / therock-10.0 headers).
DEV_INLINE void tdm_load_wpanel(const _Float16* gsrc, uint32_t lds_byte_off) {
  const uint64_t ga = (uint64_t)(uintptr_t)gsrc;
  u32x4 g0;
  g0[0] = 1u;                                        // count=1 (valid), user mode
  g0[1] = lds_byte_off;                              // lds_addr (bytes)
  g0[2] = (uint32_t)(ga & 0xFFFFFFFFu);              // global_addr[31:0]
  g0[3] = (uint32_t)((ga >> 32) & 0x01FFFFFFu)       // global_addr[56:32]
        | (2u << 30);                                // type = 2 ("image")
  i32x8 g1;
  g1[0] = (int)((1u << 16)                           // data_size = 1 -> 2 bytes
              | (1u << 20)                           // pad_enable
              | (7u << 22)                           // pad_interval: 256 DWORDs
              | (3u << 25));                         // pad_amount: 4 DWORDs
  g1[1] = (int)(512u << 16);                         // tensor_dim0[15:0]=512
  g1[2] = (int)(512u << 16);                         // dim0 hi=0 | tensor_dim1 lo=512
  g1[3] = (int)(512u << 16);                         // dim1 hi=0 | tile_dim0=512
  g1[4] = 64;                                        // tile_dim1=64, tile_dim2=0
  g1[5] = 512;                                       // tensor_dim0_stride = 512
  g1[6] = 0;
  g1[7] = 0;
  i32x4 g2 = {0, 0, 0, 0};                           // dims 3/4 unused
  i32x4 g3 = {0, 0, 0, 0};
  i32x8 g4 = {0, 0, 0, 0, 0, 0, 0, 0};               // extra group (zero-filled)
  __builtin_amdgcn_tensor_load_to_lds(g0, g1, g2, g3, g4, 0);
}
#endif

// ---------------------------------------------------------------------------
// f32 -> f16 convert (weights)
// ---------------------------------------------------------------------------
__global__ void cvt_f32_to_f16(const float* __restrict__ src, _Float16* __restrict__ dst, int n) {
  int i = blockIdx.x * blockDim.x + threadIdx.x;
  if (i < n) dst[i] = (_Float16)src[i];
}

// ---------------------------------------------------------------------------
// Pointwise conv as GEMM:  Out[o,t] = sum_c W[o,c] * In[c,t] + bias[o]
// Tile: 64 o x 128 t per workgroup (256 thr, 8 waves, wave = 16o x 64t)
// Weight panel (64 x 512 f16) TDM-loaded to LDS once; B tile staged per K step.
// IN_MODE : 0 = fp32 source, (B,C,T) layout     (x / cond)
//           1 = f16  source, (B,T,C) layout     (attention output)
// OUT_MODE: 0 = f16 out, (B,T,C) via LDS transpose   (q, k)
//           1 = f16 out, (B,C,T) direct              (v)
//           2 = f32 out, (B,C,T) direct              (final result)
// ---------------------------------------------------------------------------
template <int IN_MODE, int OUT_MODE>
__global__ void __launch_bounds__(256)
proj_gemm(const _Float16* __restrict__ Wh, const float* __restrict__ bias,
          const void* __restrict__ InP, void* __restrict__ OutP) {
  __shared__ __align__(16) _Float16 Alds[64 * ASTR];   // [o][c] padded rows
  __shared__ __align__(16) _Float16 Blds[128 * BSTR];  // [t][c] padded rows
  __shared__ __align__(16) _Float16 Olds[64 * 128];    // [o][t] (OUT_MODE 0)

  const int tid  = threadIdx.x;
  const int lane = tid & 31;
  const int wave = tid >> 5;
  const int hi   = (lane >= 16) ? 1 : 0;
  const int ln   = lane & 15;
  const int ow   = wave >> 1;   // 0..3  (o sub-tile)
  const int tw   = wave & 1;    // 0..1  (t sub-tile)

  const int tB = blockIdx.x * 128;
  const int oB = blockIdx.y * 64;
  const int b  = blockIdx.z;

#if USE_TDM
  if (wave == 0)
    tdm_load_wpanel(Wh + (size_t)oB * C_, (uint32_t)(uintptr_t)(&Alds[0]));
#else
  for (int i = 0; i < 16; ++i) {   // cooperative fallback: 64 rows x 512 halves
    const int gi   = tid + 256 * i;     // 0..4095 chunks of 8 halves
    const int row  = gi >> 6;
    const int col8 = (gi & 63) * 8;
    *(uint4*)(Alds + row * ASTR + col8) =
        *(const uint4*)(Wh + (size_t)(oB + row) * C_ + col8);
  }
#endif

  v8f acc[4];
  for (int i = 0; i < 4; ++i) acc[i] = zero8();

  for (int kk = 0; kk < C_ / 32; ++kk) {
    const int c0 = kk * 32;
    __syncthreads();
    if (IN_MODE == 0) {  // fp32 (B,C,T): read along t, convert, transpose into [t][c]
      const float* In32 = (const float*)InP;
      const int ci = tid >> 3;          // 0..31
      const int tg = (tid & 7) * 16;    // 0..112
      const float* src = In32 + (size_t)(b * C_ + c0 + ci) * T_ + tB + tg;
      for (int j = 0; j < 16; ++j)
        Blds[(tg + j) * BSTR + ci] = (_Float16)src[j];
    } else {             // f16 (B,T,C): rows already [t][c]
      const _Float16* Inh = (const _Float16*)InP;
      const int tj = tid >> 1;          // 0..127
      const int cg = (tid & 1) * 16;
      const uint4* src = (const uint4*)(Inh + (size_t)(b * T_ + tB + tj) * C_ + c0 + cg);
      uint4* dst = (uint4*)(Blds + tj * BSTR + cg);
      dst[0] = src[0];
      dst[1] = src[1];
    }
#if USE_TDM
    if (kk == 0 && wave == 0) __builtin_amdgcn_s_wait_tensorcnt(0);
#endif
    __syncthreads();

    HFrag a;  // 16 contiguous halves at K offsets {hi*8 .. +7, 16+hi*8 .. +7}
    {
      const _Float16* arow = Alds + (ow * 16 + ln) * ASTR + c0;
      a.q[0] = *(const uint4*)(arow + hi * 8);
      a.q[1] = *(const uint4*)(arow + 16 + hi * 8);
    }
    for (int nt = 0; nt < 4; ++nt) {
      HFrag bf;  // K offsets {hi*8 .. +7, 16+hi*8 .. +7}
      const _Float16* brow = Blds + (tw * 64 + nt * 16 + ln) * BSTR + hi * 8;
      bf.q[0] = *(const uint4*)(brow);
      bf.q[1] = *(const uint4*)(brow + 16);
      acc[nt] = wmma16(a.v, bf.v, acc[nt]);
    }
  }

  if (OUT_MODE == 0) {
    __syncthreads();
    for (int nt = 0; nt < 4; ++nt)
      for (int r = 0; r < 8; ++r) {
        const int o = ow * 16 + r + (hi ? 8 : 0);
        Olds[o * 128 + tw * 64 + nt * 16 + ln] = (_Float16)(acc[nt][r] + bias[oB + o]);
      }
    __syncthreads();
    _Float16* OutH = (_Float16*)OutP;
    for (int i = 0; i < 4; ++i) {
      const int it = tid + 256 * i;     // 0..1023
      const int t  = it >> 3;           // 0..127
      const int og = (it & 7) * 8;      // 0..56
      OutPk pk;
      for (int jj = 0; jj < 8; ++jj) pk.h[jj] = Olds[(og + jj) * 128 + t];
      *(uint4*)(OutH + (size_t)(b * T_ + tB + t) * C_ + oB + og) = pk.q;
    }
  } else if (OUT_MODE == 1) {
    _Float16* OutH = (_Float16*)OutP;
    for (int nt = 0; nt < 4; ++nt)
      for (int r = 0; r < 8; ++r) {
        const int o = oB + ow * 16 + r + (hi ? 8 : 0);
        const int t = tB + tw * 64 + nt * 16 + ln;
        OutH[(size_t)(b * C_ + o) * T_ + t] = (_Float16)(acc[nt][r] + bias[o]);
      }
  } else {
    float* Outf = (float*)OutP;
    for (int nt = 0; nt < 4; ++nt)
      for (int r = 0; r < 8; ++r) {
        const int o = oB + ow * 16 + r + (hi ? 8 : 0);
        const int t = tB + tw * 64 + nt * 16 + ln;
        Outf[(size_t)(b * C_ + o) * T_ + t] = acc[nt][r] + bias[o];
      }
  }
}

// ---------------------------------------------------------------------------
// Sliding-window attention (W=16, pad 8/7). One wave = 16 query positions.
// q,k in (B,T,C) f16; v in (B,C,T) f16; out attn in (B,T,C) f16.
// For t in [t0,t0+16): band j in [t0-8, t0+24), dots[t,w] = S[t, (t-t0)+w].
// ---------------------------------------------------------------------------
__global__ void __launch_bounds__(128)
local_attn(const _Float16* __restrict__ qh, const _Float16* __restrict__ kh,
           const _Float16* __restrict__ vh, _Float16* __restrict__ attn) {
  __shared__ float    Slds[4][16][32];
  __shared__ _Float16 Plds[4][16][32];

  const int tid  = threadIdx.x;
  const int lane = tid & 31;
  const int wave = tid >> 5;
  const int hi   = (lane >= 16) ? 1 : 0;
  const int ln   = lane & 15;

  const int t0 = blockIdx.x * 64 + wave * 16;
  const int h  = blockIdx.y;
  const int b  = blockIdx.z;
  const int j0 = t0 - 8;
  const size_t cbase = (size_t)h * HD_;

  // q A-fragments: rows = t, K = d (two 32-wide K steps over hd=64)
  HFrag aq[2];
  {
    const _Float16* qrow = qh + (size_t)(b * T_ + t0 + ln) * C_ + cbase;
    for (int s = 0; s < 2; ++s) {
      aq[s].q[0] = *(const uint4*)(qrow + s * 32 + hi * 8);
      aq[s].q[1] = *(const uint4*)(qrow + s * 32 + 16 + hi * 8);
    }
  }

  // S[t, j] = sum_d q[d,t] * k[d, j0+j], j-band of 32 (two 16-wide N tiles)
  v8f S[2];
  S[0] = zero8(); S[1] = zero8();
  for (int jt = 0; jt < 2; ++jt) {
    const int jpos = j0 + jt * 16 + ln;                 // this lane's N column
    const bool ok = (jpos >= 0) && (jpos < T_);
    for (int s = 0; s < 2; ++s) {
      HFrag bk;
      if (ok) {
        const _Float16* krow =
            kh + (size_t)(b * T_ + jpos) * C_ + cbase + s * 32 + (hi ? 8 : 0);
        bk.q[0] = *(const uint4*)(krow);
        bk.q[1] = *(const uint4*)(krow + 16);
      } else {
        for (int p = 0; p < 8; ++p) bk.u[p] = 0u;       // zero-pad => dot = 0
      }
      S[jt] = wmma16(aq[s].v, bk.v, S[jt]);
    }
  }

  for (int jt = 0; jt < 2; ++jt)
    for (int r = 0; r < 8; ++r)
      Slds[wave][r + (hi ? 8 : 0)][jt * 16 + ln] = S[jt][r];
  __syncthreads();

  // Softmax over the 16-wide diagonal band; build P (16 x 32, zero off-band)
  if (lane < 16) {
    const int m = lane;
    float pv[16], vmax = -1e30f;
    for (int wd = 0; wd < 16; ++wd) {
      pv[wd] = Slds[wave][m][m + wd] * SCALE;
      vmax = fmaxf(vmax, pv[wd]);
    }
    float sum = 0.0f;
    for (int wd = 0; wd < 16; ++wd) { pv[wd] = __expf(pv[wd] - vmax); sum += pv[wd]; }
    const float inv = 1.0f / sum;
    for (int col = 0; col < 32; ++col) {
      const int wd = col - m;
      const float p = (wd >= 0 && wd < 16) ? pv[wd] * inv : 0.0f;
      Plds[wave][m][col] = (_Float16)p;
    }
  }
  __syncthreads();

  // O[t, d] = sum_j P[t, j] * v[d, j0+j]   (K=32 band in one WMMA per d-chunk)
  HFrag ap;
  {
    const _Float16* prow = &Plds[wave][ln][0];
    ap.q[0] = *(const uint4*)(prow + hi * 8);
    ap.q[1] = *(const uint4*)(prow + 16 + hi * 8);
  }

  for (int ch = 0; ch < 4; ++ch) {
    HFrag bv;
    const size_t vrow = ((size_t)(b * C_) + cbase + ch * 16 + ln) * T_;
    for (int e = 0; e < 16; ++e) {
      const int jpos = j0 + (e < 8 ? (hi ? 8 : 0) + e : 16 + (hi ? 8 : 0) + (e - 8));
      bv.h[e] = (jpos >= 0 && jpos < T_) ? vh[vrow + jpos] : (_Float16)0.0f;
    }
    v8f o = wmma16(ap.v, bv.v, zero8());
    for (int r = 0; r < 8; ++r) {
      const int t = t0 + r + (hi ? 8 : 0);
      attn[(size_t)(b * T_ + t) * C_ + cbase + ch * 16 + ln] = (_Float16)o[r];
    }
  }
}

}  // namespace

extern "C" void kernel_launch(void* const* d_in, const int* in_sizes, int n_in,
                              void* d_out, int out_size, void* d_ws, size_t ws_size,
                              hipStream_t stream) {
  (void)in_sizes; (void)n_in; (void)out_size; (void)ws_size;
  const float* x    = (const float*)d_in[0];
  const float* cond = (const float*)d_in[1];
  const float* Wq   = (const float*)d_in[2];
  const float* bq   = (const float*)d_in[3];
  const float* Wk   = (const float*)d_in[4];
  const float* bk   = (const float*)d_in[5];
  const float* Wv   = (const float*)d_in[6];
  const float* bv   = (const float*)d_in[7];
  const float* Wo   = (const float*)d_in[8];
  const float* bo   = (const float*)d_in[9];

  _Float16* ws = (_Float16*)d_ws;
  const size_t WSZ = (size_t)C_ * C_;
  const size_t BCT = (size_t)B_ * C_ * T_;
  _Float16* wqh  = ws;
  _Float16* wkh  = wqh + WSZ;
  _Float16* wvh  = wkh + WSZ;
  _Float16* woh  = wvh + WSZ;
  _Float16* qbuf = woh + WSZ;        // (B,T,C)
  _Float16* kbuf = qbuf + BCT;       // (B,T,C)
  _Float16* vbuf = kbuf + BCT;       // (B,C,T)
  _Float16* abuf = vbuf + BCT;       // (B,T,C)

  const int nW = C_ * C_;
  dim3 cgrid((nW + 255) / 256), cblk(256);
  cvt_f32_to_f16<<<cgrid, cblk, 0, stream>>>(Wq, wqh, nW);
  cvt_f32_to_f16<<<cgrid, cblk, 0, stream>>>(Wk, wkh, nW);
  cvt_f32_to_f16<<<cgrid, cblk, 0, stream>>>(Wv, wvh, nW);
  cvt_f32_to_f16<<<cgrid, cblk, 0, stream>>>(Wo, woh, nW);

  dim3 ggrid(T_ / 128, C_ / 64, B_), gblk(256);
  proj_gemm<0, 0><<<ggrid, gblk, 0, stream>>>(wqh, bq, (const void*)x,    (void*)qbuf);
  proj_gemm<0, 0><<<ggrid, gblk, 0, stream>>>(wkh, bk, (const void*)cond, (void*)kbuf);
  proj_gemm<0, 1><<<ggrid, gblk, 0, stream>>>(wvh, bv, (const void*)cond, (void*)vbuf);

  dim3 agrid(T_ / 64, H_, B_), ablk(128);
  local_attn<<<agrid, ablk, 0, stream>>>(qbuf, kbuf, vbuf, abuf);

  proj_gemm<1, 2><<<ggrid, gblk, 0, stream>>>(woh, bo, (const void*)abuf, d_out);
}
